// Model_48816598286940
// MI455X (gfx1250) — matrix operand order repaired and checked by reading, not compiled, not run on hardware
//
#include <hip/hip_runtime.h>
#include <hip/hip_bf16.h>

// ---------------------------------------------------------------------------
// bf16 WMMA GEMM pipeline for the 12-layer GCN encoder/decoder.
//   proj:  H = X @ W + b          (stored transposed, bf16)
//   adj :  Y = relu(DAD @ H)      (per batch, bf16; final layer f32 -> d_out)
// Each wave computes a 32x32 output tile = 2x2 grid of 16x16 WMMA tiles,
// via v_wmma_f32_16x16x32_bf16 (K step 32). Branchless inner loop: lanes with
// out-of-range output columns load a clamped valid row of B^T and compute
// garbage columns that are never stored (WMMA columns are independent).
// Adjacency kernel stages its shared DAD row-panel (16 KB) in LDS via
// GLOBAL_LOAD_ASYNC_TO_LDS_B128 (ASYNCcnt) when available.
// Node dim padded 255 -> 256 (zero-padded DAD rows/cols keep padding inert).
// ---------------------------------------------------------------------------

typedef __attribute__((ext_vector_type(16))) __bf16 bf16x16;
typedef __attribute__((ext_vector_type(8)))  __bf16 bf16x8;
typedef __attribute__((ext_vector_type(8)))  float  v8f;

#define GWAVES 4   // waves per block in the GEMM kernels

#if __has_builtin(__builtin_amdgcn_global_load_async_to_lds_b128)
#define HAVE_ASYNC_LDS 1
typedef int v4i __attribute__((vector_size(16)));
typedef __attribute__((address_space(1))) v4i gv4i;   // global int4
typedef __attribute__((address_space(3))) v4i lv4i;   // LDS int4
#else
#define HAVE_ASYNC_LDS 0
#endif

__device__ __forceinline__ void wait_async0() {
#if HAVE_ASYNC_LDS
#if __has_builtin(__builtin_amdgcn_s_wait_asynccnt)
  __builtin_amdgcn_s_wait_asynccnt(0);
#else
  asm volatile("s_wait_asynccnt 0" ::: "memory");
#endif
#endif
}

// A fragment (16x32 bf16): lane half h -> element e = K 16h+e:
// one contiguous 32B run at p (split into two 16B loads).
__device__ __forceinline__ bf16x16 ld_frag_a(const __bf16* p) {
  bf16x8 lo = *(const bf16x8*)(p);
  bf16x8 hi = *(const bf16x8*)(p + 8);
  return __builtin_shufflevector(lo, hi,
      0, 1, 2, 3, 4, 5, 6, 7, 8, 9, 10, 11, 12, 13, 14, 15);
}

// B fragment (32x16 bf16): lane holds one column, element e -> K = 16h + e:
// one contiguous 32B run at p (split into two 16B loads).
__device__ __forceinline__ bf16x16 ld_frag_b(const __bf16* p) {
  bf16x8 lo = *(const bf16x8*)(p);
  bf16x8 hi = *(const bf16x8*)(p + 8);
  return __builtin_shufflevector(lo, hi,
      0, 1, 2, 3, 4, 5, 6, 7, 8, 9, 10, 11, 12, 13, 14, 15);
}

// 32x32 wave tile: acc[0]=C(m0,n0) acc[1]=C(m0,n1) acc[2]=C(m1,n0) acc[3]=C(m1,n1)
// A0/A1 may point to LDS (adjacency) or global (projection); inlined per site.
__device__ __forceinline__ void gemm_2x2(
    const __bf16* A0, const __bf16* A1,
    const __bf16* B0, const __bf16* B1,
    int K, int hf, v8f acc[4]) {
  const int ao = 16 * hf;
  const int bo = 16 * hf;
  #pragma unroll 2
  for (int k0 = 0; k0 < K; k0 += 32) {
    bf16x16 a0 = ld_frag_a(A0 + k0 + ao);
    bf16x16 a1 = ld_frag_a(A1 + k0 + ao);
    bf16x16 b0 = ld_frag_b(B0 + k0 + bo);
    bf16x16 b1 = ld_frag_b(B1 + k0 + bo);
    acc[0] = __builtin_amdgcn_wmma_f32_16x16x32_bf16(false, a0, false, b0, (short)0, acc[0], false, false);
    acc[1] = __builtin_amdgcn_wmma_f32_16x16x32_bf16(false, a0, false, b1, (short)0, acc[1], false, false);
    acc[2] = __builtin_amdgcn_wmma_f32_16x16x32_bf16(false, a1, false, b0, (short)0, acc[2], false, false);
    acc[3] = __builtin_amdgcn_wmma_f32_16x16x32_bf16(false, a1, false, b1, (short)0, acc[3], false, false);
  }
}

// ---- projection: Ht[b][d][node] = (X @ W)[b*256+node][d] + bias[d] ---------
// X : bf16 [65536 x K] row-major, WT : bf16 [N x K] (= W transposed)
__global__ __launch_bounds__(128) void proj_kernel(
    const __bf16* __restrict__ X, const __bf16* __restrict__ WT,
    const float* __restrict__ bias, __bf16* __restrict__ Ht,
    int N, int K) {
  const int wid  = threadIdx.x >> 5;
  const int lane = threadIdx.x & 31;
  const int hf   = lane >> 4;
  const int lidx = lane & 15;
  const int mT = 65536 >> 5;                // 2048 row tiles of 32
  const int nT = (N + 31) >> 5;
  const long t = (long)blockIdx.x * GWAVES + wid;
  if (t >= (long)mT * nT) return;           // wave-uniform -> EXEC all-1s
  const int tm = (int)(t % mT);
  const int tn = (int)(t / mT);

  const int row0 = tm * 32 + lidx;          // global rows of X (this lane)
  const int row1 = row0 + 16;
  const int col0 = tn * 32 + lidx;          // output features (this lane)
  const int col1 = col0 + 16;
  const int c0 = col0 < N ? col0 : 0;       // clamp: garbage cols never stored
  const int c1 = col1 < N ? col1 : 0;

  const __bf16* A0 = X + (size_t)row0 * K;
  const __bf16* A1 = X + (size_t)row1 * K;
  const __bf16* B0 = WT + (size_t)c0 * K;
  const __bf16* B1 = WT + (size_t)c1 * K;
  __builtin_prefetch(A0, 0, 0);
  __builtin_prefetch(A1, 0, 0);
  __builtin_prefetch(B0, 0, 0);
  __builtin_prefetch(B1, 0, 0);

  v8f acc[4] = {{}, {}, {}, {}};
  gemm_2x2(A0, A1, B0, B1, K, hf, acc);

  const int gm0 = tm * 32;
  const int bat = gm0 >> 8;                 // 256 nodes/batch, 32 | 256
  #pragma unroll
  for (int s = 0; s < 4; ++s) {
    const int col = (s & 1) ? col1 : col0;
    if (col < N) {
      const float bv = bias[col];
      bf16x8 o;
      #pragma unroll
      for (int r = 0; r < 8; ++r) o[r] = (__bf16)(acc[s][r] + bv);
      const int m0 = (gm0 & 255) + (s >> 1) * 16 + hf * 8;
      // transposed, packed store: 8 consecutive nodes -> one 16B store
      *(bf16x8*)(Ht + ((size_t)bat * N + col) * 256 + m0) = o;
    }
  }
}

// ---- adjacency: Out[b][node][d] = relu(DAD @ H_b)[node][d] ------------------
// DAD: bf16 [256 x 256]; Ht: bf16 [B][N][256] (H transposed per batch)
// One block per (batch, m-tile): the 16 KB DAD row-panel is staged in LDS
// (async DMA when available) and shared by the 4 waves striding over n-tiles.
template <bool OUT_F32>
__global__ __launch_bounds__(128) void adj_kernel(
    const __bf16* __restrict__ DAD, const __bf16* __restrict__ Ht,
    void* __restrict__ Out, int N) {
  __shared__ __bf16 As[32 * 256];           // 16 KB panel: rows tm*32..tm*32+31

  const int wid  = threadIdx.x >> 5;
  const int lane = threadIdx.x & 31;
  const int hf   = lane >> 4;
  const int lidx = lane & 15;
  const int bat = blockIdx.x >> 3;          // 256 batches
  const int tm  = blockIdx.x & 7;           // 8 row tiles of 32

  // ---- stage contiguous 16 KB DAD panel into LDS (1024 x 16B chunks) ----
  const __bf16* gpan = DAD + (size_t)tm * 32 * 256;
#if HAVE_ASYNC_LDS
  #pragma unroll
  for (int it = 0; it < 8; ++it) {
    const int c = it * 128 + threadIdx.x;   // chunk index
    __builtin_amdgcn_global_load_async_to_lds_b128(
        (gv4i*)(gpan + (size_t)c * 8), (lv4i*)(As + (size_t)c * 8), 0, 0);
  }
  wait_async0();
#else
  #pragma unroll
  for (int it = 0; it < 8; ++it) {
    const int c = it * 128 + threadIdx.x;
    *(bf16x8*)(As + (size_t)c * 8) = *(const bf16x8*)(gpan + (size_t)c * 8);
  }
#endif
  __syncthreads();

  const int nT = (N + 31) >> 5;
  const __bf16* A0 = As + (size_t)lidx * 256;        // LDS fragment rows
  const __bf16* A1 = As + (size_t)(lidx + 16) * 256;
  const __bf16* Btb = Ht + (size_t)bat * N * 256;

  for (int tn = wid; tn < nT; tn += GWAVES) {        // wave-uniform loop
    const int col0 = tn * 32 + lidx;
    const int col1 = col0 + 16;
    const int c0 = col0 < N ? col0 : 0;
    const int c1 = col1 < N ? col1 : 0;
    const __bf16* B0 = Btb + (size_t)c0 * 256;
    const __bf16* B1 = Btb + (size_t)c1 * 256;
    __builtin_prefetch(B0, 0, 0);
    __builtin_prefetch(B1, 0, 0);

    v8f acc[4] = {{}, {}, {}, {}};
    gemm_2x2(A0, A1, B0, B1, 256, hf, acc);

    #pragma unroll
    for (int s = 0; s < 4; ++s) {
      const int col = (s & 1) ? col1 : col0;
      const int mbase = tm * 32 + (s >> 1) * 16 + hf * 8;
      if (OUT_F32) {
        if (col < 2) {
          float* O = (float*)Out;           // [B][255][2]
          #pragma unroll
          for (int r = 0; r < 8; ++r) {
            const int m = mbase + r;
            if (m < 255) O[((size_t)bat * 255 + m) * 2 + col] = fmaxf(acc[s][r], 0.f);
          }
        }
      } else {
        if (col < N) {
          __bf16* O = (__bf16*)Out;         // [B][256][N] = next layer's X
          #pragma unroll
          for (int r = 0; r < 8; ++r) {
            const int m = mbase + r;
            O[((size_t)bat * 256 + m) * N + col] = (__bf16)fmaxf(acc[s][r], 0.f);
          }
        }
      }
    }
  }
}

// ---- converters -------------------------------------------------------------
__global__ void cvt_H(const float* __restrict__ H, __bf16* __restrict__ X) {
  // H [256,255,64] f32 -> X [256,256,64] bf16, pad node 255 with zeros
  const size_t i = (size_t)blockIdx.x * blockDim.x + threadIdx.x;
  if (i >= (size_t)256 * 256 * 64) return;
  const int bat  = (int)(i / (256 * 64));
  const int rem  = (int)(i % (256 * 64));
  const int node = rem / 64;
  const int f    = rem % 64;
  X[i] = (node < 255) ? (__bf16)H[((size_t)bat * 255 + node) * 64 + f]
                      : (__bf16)0.0f;
}

__global__ void cvt_WT(const float* __restrict__ W, __bf16* __restrict__ WT,
                       int Din, int Dout) {
  // W [Din,Dout] f32 -> WT [Dout,Din] bf16
  const int i = blockIdx.x * blockDim.x + threadIdx.x;
  if (i >= Din * Dout) return;
  const int dout = i / Din;
  const int din  = i % Din;
  WT[i] = (__bf16)W[(size_t)din * Dout + dout];
}

__global__ void cvt_DAD(const float* __restrict__ S, __bf16* __restrict__ D) {
  // [255,255] f32 -> [256,256] bf16, zero-padded
  const int i = blockIdx.x * blockDim.x + threadIdx.x;
  if (i >= 256 * 256) return;
  const int r = i >> 8, c = i & 255;
  D[i] = (r < 255 && c < 255) ? (__bf16)S[(size_t)r * 255 + c] : (__bf16)0.0f;
}

// ---------------------------------------------------------------------------
extern "C" void kernel_launch(void* const* d_in, const int* in_sizes, int n_in,
                              void* d_out, int out_size, void* d_ws, size_t ws_size,
                              hipStream_t stream) {
  (void)in_sizes; (void)n_in; (void)out_size; (void)ws_size;

  static const int DIN[12]  = {64, 64, 64, 128, 128, 256, 256, 256, 256, 128, 128, 64};
  static const int DOUT[12] = {64, 64, 128, 128, 256, 256, 256, 256, 128, 128, 64, 2};

  const float* H = (const float*)d_in[0];
  const float* dads[4] = {(const float*)d_in[1], (const float*)d_in[2],
                          (const float*)d_in[3], (const float*)d_in[4]};
  const float* Wp[12];
  const float* Bp[12];
  for (int i = 0; i < 6; ++i) {
    Wp[i]     = (const float*)d_in[5 + 4 * i];     // enc_w{i}
    Bp[i]     = (const float*)d_in[6 + 4 * i];     // enc_b{i}
    Wp[6 + i] = (const float*)d_in[7 + 4 * i];     // dec_w{i}
    Bp[6 + i] = (const float*)d_in[8 + 4 * i];     // dec_b{i}
  }

  // workspace carve-up
  char* ws = (char*)d_ws;
  const size_t XBF_BYTES = (size_t)256 * 256 * 256 * 2;   // 33,554,432
  __bf16* Xbf  = (__bf16*)(ws);
  __bf16* Htb  = (__bf16*)(ws + XBF_BYTES);
  __bf16* DADb = (__bf16*)(ws + 2 * XBF_BYTES);
  __bf16* WTb  = (__bf16*)(ws + 2 * XBF_BYTES + (size_t)4 * 65536 * 2);

  // --- conversions (deterministic, re-run every call) ---
  {
    const size_t total = (size_t)256 * 256 * 64;
    cvt_H<<<(unsigned)((total + 255) / 256), 256, 0, stream>>>(H, Xbf);
  }
  for (int j = 0; j < 4; ++j)
    cvt_DAD<<<(65536 + 255) / 256, 256, 0, stream>>>(dads[j], DADb + (size_t)j * 65536);

  __bf16* WT[12];
  size_t woff = 0;
  for (int L = 0; L < 12; ++L) {
    WT[L] = WTb + woff;
    const int cnt = DIN[L] * DOUT[L];
    cvt_WT<<<(cnt + 255) / 256, 256, 0, stream>>>(Wp[L], WT[L], DIN[L], DOUT[L]);
    woff += (size_t)cnt;
  }

  // --- 12 layers ---
  for (int L = 0; L < 12; ++L) {
    const int K = DIN[L], N = DOUT[L];
    const int nT = (N + 31) >> 5;

    // projection: Xbf [65536 x K] @ WT -> Htb (transposed, +bias)
    {
      const long tiles = 2048L * nT;
      const unsigned blocks = (unsigned)((tiles + GWAVES - 1) / GWAVES);
      proj_kernel<<<blocks, 128, 0, stream>>>(Xbf, WT[L], Bp[L], Htb, N, K);
    }

    // adjacency: relu(DAD @ H_b) per batch; 1 block per (batch, m-tile)
    const int dsel = (L < 6) ? ((L & 1) ? 1 : 0) : (((L - 6) & 1) ? 3 : 2);
    const __bf16* dad = DADb + (size_t)dsel * 65536;
    if (L == 11)
      adj_kernel<true><<<2048, 128, 0, stream>>>(dad, Htb, d_out, N);
    else
      adj_kernel<false><<<2048, 128, 0, stream>>>(dad, Htb, Xbf, N);
  }
}
